// RestGCNEqualHidden_11321533792498
// MI455X (gfx1250) — matrix operand order repaired and checked by reading, not compiled, not run on hardware
//
#include <hip/hip_runtime.h>
#include <hip/hip_bf16.h>

#define H 128

typedef __attribute__((ext_vector_type(16))) __bf16 v16bf;
typedef __attribute__((ext_vector_type(8)))  float  v8f;
typedef __attribute__((ext_vector_type(4)))  unsigned int u32x4;
typedef __attribute__((__vector_size__(16))) int gv4i;   // matches builtin param type

#define LDS_AS __attribute__((address_space(3)))
#define GLB_AS __attribute__((address_space(1)))

union FragB16 {
  v16bf v;
  u32x4 q[2];
};

// f32 -> bf16 round-to-nearest-even (bit manipulation; matches hardware RNE)
__device__ __forceinline__ unsigned short f2bf(float f) {
  unsigned int u = __float_as_uint(f);
  u += 0x7FFFu + ((u >> 16) & 1u);
  return (unsigned short)(u >> 16);
}

// ---------------------------------------------------------------------------
// Prep: W (128x128 f32, K-major) -> Wt (128x128 bf16, N-major / transposed)
// so a B-fragment is two contiguous 16B reads per lane.
// ---------------------------------------------------------------------------
__global__ void cvt_w_kernel(const float* __restrict__ W,
                             unsigned short* __restrict__ Wt) {
  int idx = blockIdx.x * blockDim.x + threadIdx.x;   // 0..16383
  int n = idx >> 7;
  int k = idx & 127;
  Wt[n * H + k] = f2bf(W[k * H + n]);
}

// ---------------------------------------------------------------------------
// Degree / normalization: dis[i] = rsqrt(1 + indegree(i))
// ---------------------------------------------------------------------------
__global__ void deg_init_kernel(float* __restrict__ deg, int n) {
  int i = blockIdx.x * blockDim.x + threadIdx.x;
  if (i < n) deg[i] = 1.0f;   // self loop
}

__global__ void deg_acc_kernel(const int* __restrict__ dst,
                               float* __restrict__ deg, int E) {
  int e = blockIdx.x * blockDim.x + threadIdx.x;
  if (e < E) atomicAdd(&deg[dst[e]], 1.0f);
}

__global__ void deg_rsqrt_kernel(float* __restrict__ deg, int n) {
  int i = blockIdx.x * blockDim.x + threadIdx.x;
  if (i < n) deg[i] = rsqrtf(fmaxf(deg[i], 1.0f));
}

// ---------------------------------------------------------------------------
// GEMM: t[N x 128] = (inA [+ inB]) @ W, via v_wmma_f32_16x16x32_bf16.
// Block = 256 threads (8 waves), 128 rows per block, full 128 cols.
//  - Input tile staged into LDS as bf16 (padded rows).
//  - Wt (already bf16) staged into LDS via GLOBAL_LOAD_ASYNC_TO_LDS_B128
//    (ASYNCcnt path), hidden under the A-tile conversion work.
// ---------------------------------------------------------------------------
__global__ __launch_bounds__(256) void gemm_bf16_kernel(
    const float* __restrict__ inA, const float* __restrict__ inB,
    const unsigned short* __restrict__ Wt, float* __restrict__ out, int nRows) {
  __shared__ unsigned short sA[128][136];   // 272B row pitch (16B aligned, padded)
  __shared__ unsigned short sW[128][136];   // Wt tile, same pitch

  const int tid = threadIdx.x;
  const int rowBase = blockIdx.x * 128;

  // ---- issue async staging of Wt (bf16, no conversion needed) ----
  // 128 rows x 16 chunks of 16B = 2048 chunks; 8 per thread.
  for (int i = 0; i < 8; ++i) {
    int chunk = tid + i * 256;            // 0..2047
    int row = chunk >> 4;
    int c8  = (chunk & 15) * 8;           // ushort offset within row
    const unsigned short* gsrc = Wt + row * H + c8;
    unsigned short* ldst = &sW[row][c8];
#if __has_builtin(__builtin_amdgcn_global_load_async_to_lds_b128)
    __builtin_amdgcn_global_load_async_to_lds_b128(
        (GLB_AS gv4i*)gsrc, (LDS_AS gv4i*)ldst, 0, 0);
#else
    *(u32x4*)ldst = *(const u32x4*)gsrc;
#endif
  }

  // ---- stage 128x128 f32 tile (optionally sum of two inputs) -> bf16 LDS ----
  for (int it = 0; it < 16; ++it) {
    int idx = tid + it * 256;          // float4 index, 0..4095
    int r  = idx >> 5;                 // row in tile
    int c4 = idx & 31;                 // float4 column
    int gr = rowBase + r;
    int sr = gr < nRows ? gr : (nRows - 1);   // clamp tail reads
    float4 v = ((const float4*)(inA + (size_t)sr * H))[c4];
    if (inB) {
      float4 w = ((const float4*)(inB + (size_t)sr * H))[c4];
      v.x += w.x; v.y += w.y; v.z += w.z; v.w += w.w;
    }
    int c = c4 * 4;
    sA[r][c + 0] = f2bf(v.x);
    sA[r][c + 1] = f2bf(v.y);
    sA[r][c + 2] = f2bf(v.z);
    sA[r][c + 3] = f2bf(v.w);
  }

#if __has_builtin(__builtin_amdgcn_global_load_async_to_lds_b128)
#if __has_builtin(__builtin_amdgcn_s_wait_asynccnt)
  __builtin_amdgcn_s_wait_asynccnt(0);
#else
  asm volatile("s_wait_asynccnt 0" ::: "memory");
#endif
#endif
  __syncthreads();

  const int wave = tid >> 5;
  const int lane = tid & 31;
  const int nl   = lane & 15;   // 0..15: M index (A) / N index (B, C/D)
  const int half = lane >> 4;   // lane group
  const int mrow = wave * 16;   // this wave's row strip within the tile

  v8f acc[8];
  for (int nt = 0; nt < 8; ++nt)
    for (int i = 0; i < 8; ++i) acc[nt][i] = 0.0f;

  // ---- K loop: 128 = 4 chunks of 32 ----
  for (int kc = 0; kc < 4; ++kc) {
    // A fragment (16x32 bf16): lane group h holds K = kc*32 + {h*8..h*8+7}
    // in u[0..7] and K = kc*32 + 16 + {h*8..h*8+7} in u[8..15]   (ISA 7.12.2)
    FragB16 a;
    const unsigned short* rowp = &sA[mrow + nl][kc * 32];
    a.q[0] = *(const u32x4*)(rowp + half * 8);
    a.q[1] = *(const u32x4*)(rowp + 16 + half * 8);

    for (int nt = 0; nt < 8; ++nt) {
      // B fragment (32x16 bf16): lane group h holds K = kc*32 + {h*8..h*8+7}
      // in u[0..7] and K = kc*32 + 16 + {h*8..h*8+7} in u[8..15]. From the LDS-staged Wt tile.
      FragB16 b;
      const unsigned short* wp = &sW[nt * 16 + nl][kc * 32 + half * 8];
      b.q[0] = *(const u32x4*)(wp);
      b.q[1] = *(const u32x4*)(wp + 16);

      acc[nt] = __builtin_amdgcn_wmma_f32_16x16x32_bf16(
          /*neg_a=*/false, a.v, /*neg_b=*/false, b.v,
          /*c_mod=*/(short)0, acc[nt], /*reuse_a=*/false, /*reuse_b=*/false);
    }
  }

  // ---- store: C/D layout: VGPR r -> M = half*8 + r, N = nl (per n-tile) ----
  for (int nt = 0; nt < 8; ++nt) {
    int col = nt * 16 + nl;
    for (int r = 0; r < 8; ++r) {
      int row = rowBase + mrow + half * 8 + r;
      if (row < nRows) out[(size_t)row * H + col] = acc[nt][r];
    }
  }
}

// ---------------------------------------------------------------------------
// dest[i][c] = bias[c] + t[i][c] * dis[i]^2   (bias + self-loop term fused)
// ---------------------------------------------------------------------------
__global__ void init_dest_kernel(float* __restrict__ dest,
                                 const float* __restrict__ t,
                                 const float* __restrict__ dis,
                                 const float* __restrict__ bias, int n) {
  int idx = blockIdx.x * blockDim.x + threadIdx.x;
  int i = idx >> 7;
  int c = idx & 127;
  if (i < n) {
    float d = dis[i];
    dest[idx] = bias[c] + t[idx] * d * d;
  }
}

// ---------------------------------------------------------------------------
// Edge scatter: one wave per edge; lane handles a float4 of the row.
// Gather is a fully coalesced 512B row read; scatter via f32 atomics (L2).
// ---------------------------------------------------------------------------
__global__ __launch_bounds__(256) void scatter_kernel(
    const float* __restrict__ t, const int* __restrict__ src,
    const int* __restrict__ dst, const float* __restrict__ dis,
    float* __restrict__ dest, int E) {
  int e = blockIdx.x * 8 + (threadIdx.x >> 5);
  if (e >= E) return;
  int lane = threadIdx.x & 31;
  int s = src[e];
  int d = dst[e];
  float nrm = dis[s] * dis[d];
  float4 v = ((const float4*)(t + (size_t)s * H))[lane];
  float* dp = dest + (size_t)d * H + lane * 4;
  atomicAdd(dp + 0, v.x * nrm);
  atomicAdd(dp + 1, v.y * nrm);
  atomicAdd(dp + 2, v.z * nrm);
  atomicAdd(dp + 3, v.w * nrm);
}

__global__ void relu_kernel(float* __restrict__ p, int n) {
  int i = blockIdx.x * blockDim.x + threadIdx.x;
  if (i < n) p[i] = fmaxf(p[i], 0.0f);
}

// ---------------------------------------------------------------------------
// Host side
// ---------------------------------------------------------------------------
extern "C" void kernel_launch(void* const* d_in, const int* in_sizes, int n_in,
                              void* d_out, int out_size, void* d_ws,
                              size_t ws_size, hipStream_t stream) {
  const float* x  = (const float*)d_in[0];
  const int*   ei = (const int*)d_in[1];
  const float* W1 = (const float*)d_in[2];
  const float* b1 = (const float*)d_in[3];
  const float* W2 = (const float*)d_in[4];
  const float* b2 = (const float*)d_in[5];
  const float* W3 = (const float*)d_in[6];
  const float* b3 = (const float*)d_in[7];
  const float* W4 = (const float*)d_in[8];
  const float* b4 = (const float*)d_in[9];

  const int N = in_sizes[0] / H;      // 100000
  const int E = in_sizes[1] / 2;      // 1600000
  const int* src = ei;
  const int* dstp = ei + E;

  // ---- workspace carve-up ----
  char* ws = (char*)d_ws;
  auto take = [&](size_t bytes) {
    char* p = ws;
    ws += (bytes + 255) & ~(size_t)255;
    return p;
  };
  const size_t NH = (size_t)N * H;
  float* dis = (float*)take((size_t)N * sizeof(float));
  unsigned short* Wt1 = (unsigned short*)take(H * H * sizeof(unsigned short));
  unsigned short* Wt2 = (unsigned short*)take(H * H * sizeof(unsigned short));
  unsigned short* Wt3 = (unsigned short*)take(H * H * sizeof(unsigned short));
  unsigned short* Wt4 = (unsigned short*)take(H * H * sizeof(unsigned short));
  float* t    = (float*)take(NH * sizeof(float));
  float* bufA = (float*)take(NH * sizeof(float));   // x1, later x4
  float* bufB = (float*)take(NH * sizeof(float));   // x2
  float* bufC = (float*)take(NH * sizeof(float));   // x3
  float* outp = (float*)d_out;

  // ---- prep: bf16 transposed weights + GCN normalization ----
  cvt_w_kernel<<<(H * H) / 256, 256, 0, stream>>>(W1, Wt1);
  cvt_w_kernel<<<(H * H) / 256, 256, 0, stream>>>(W2, Wt2);
  cvt_w_kernel<<<(H * H) / 256, 256, 0, stream>>>(W3, Wt3);
  cvt_w_kernel<<<(H * H) / 256, 256, 0, stream>>>(W4, Wt4);
  deg_init_kernel<<<(N + 255) / 256, 256, 0, stream>>>(dis, N);
  deg_acc_kernel<<<(E + 255) / 256, 256, 0, stream>>>(dstp, dis, E);
  deg_rsqrt_kernel<<<(N + 255) / 256, 256, 0, stream>>>(dis, N);

  const int gemmBlocks = (N + 127) / 128;
  const int tot = N * H;
  auto conv = [&](const float* inA, const float* inB,
                  const unsigned short* Wt, const float* bias, float* dest) {
    gemm_bf16_kernel<<<gemmBlocks, 256, 0, stream>>>(inA, inB, Wt, t, N);
    init_dest_kernel<<<(tot + 255) / 256, 256, 0, stream>>>(dest, t, dis, bias, N);
    scatter_kernel<<<(E + 7) / 8, 256, 0, stream>>>(t, src, dstp, dis, dest, E);
    relu_kernel<<<(tot + 255) / 256, 256, 0, stream>>>(dest, tot);
  };

  conv(x,    nullptr, Wt1, b1, bufA);   // x1 = relu(conv(x))
  conv(bufA, nullptr, Wt2, b2, bufB);   // x2 = relu(conv(x1))
  conv(bufB, bufA,    Wt3, b3, bufC);   // x3 = relu(conv(x2 + x1))
  conv(bufB, bufC,    Wt4, b4, bufA);   // x4 = relu(conv(x2 + x3))  (reuse x1 buf)
  conv(bufA, bufC,    Wt4, b4, outp);   // out = relu(conv(x4 + x3)) (conv4 reused)
}
